// Transformer_Retrieval_79946521247867
// MI455X (gfx1250) — matrix operand order repaired and checked by reading, not compiled, not run on hardware
//
#include <hip/hip_runtime.h>
#include <hip/hip_bf16.h>
#include <cstdint>
#include <cstddef>

// ----------------------------------------------------------------------------
// d_in flattening assumption (depth-first, dict insertion order from setup_inputs):
//  0: src (256,14,1024)
//  1..5: tokens ss, ms, ls, high, low (1024 each)
//  6: a_fmri (1)
//  7..24: embed: dorsal.w/b, ventral.w/b, visual.w/b, faces.w/b, words.w/b,
//         places.w/b, bodies.w/b, small_w(14,1024,1024), small_b(14,1024), ln.g, ln.b
//  25 + 40*l .. : per layer l in 0..3:
//    +0..19 : msa[0..4] each {w_in(3072,1024), b_in, w_out(1024,1024), b_out}
//    +20,21 : n11.g/b    +22..25: ffn1 l1.w/b l2.w/b    +26,27: n12.g/b
//    +28..31: cross w_in/b_in/w_out/b_out
//    +32,33 : n21.g/b    +34..37: ffn2 l1.w/b l2.w/b    +38,39: n22.g/b
//  185..188: gate l1.w/b l2.w/b
//  189,190: high_fc w/b   191,192: low_fc w/b
//  193..196: cate l1.w/b l2.w/b   197..200: name l1.w/b l2.w/b
// Outputs (concat): high(256*768), low(256*768), cate_feat(256*12),
//                   cate_prob(256*12), name_out(256*80)
// ----------------------------------------------------------------------------

#define DEVI __device__ __forceinline__

typedef __attribute__((ext_vector_type(16))) __bf16 bf16x16;
typedef __attribute__((ext_vector_type(2)))  __bf16 bf16x2;
typedef __attribute__((ext_vector_type(8)))  float  f32x8;
typedef __attribute__((ext_vector_type(4)))  float  f32x4;
typedef __attribute__((ext_vector_type(2)))  float  f32x2;
typedef __attribute__((ext_vector_type(2)))  unsigned int u32x2;

union BFrag { bf16x16 v; unsigned short u[16]; };

// Native converts -> v_cvt_(pk_)bf16_f32 (RNE).
DEVI unsigned short f2bf(float f) {
  union { __bf16 h; unsigned short s; } u; u.h = (__bf16)f; return u.s;
}
// Packed convert: one v_cvt_pk_bf16_f32 per pair.
DEVI unsigned int pack2bf(float a, float b) {
  union { bf16x2 v; unsigned u; } r;
#if __has_builtin(__builtin_amdgcn_cvt_pk_bf16_f32)
  r.v = __builtin_amdgcn_cvt_pk_bf16_f32(a, b);
#else
  r.v = __builtin_convertvector((f32x2){a, b}, bf16x2);
#endif
  return r.u;
}

DEVI float act_apply(float v, int act) {
  if (act == 1) return v > 0.f ? v : 0.f;
  if (act == 2) return 1.f / (1.f + __expf(-v));
  return v;
}

// ---------------------------------------------------------------------------
// Fast GEMM: C[M,N] = act(A[M,K] @ W[N,K]^T + bias[N])
// Requirements: M%128==0, N%128==0, K%32==0 (host guarantees).
// Block tile 128x128, K-step 32. 8 waves: wm=wave&3 (32 rows), wn=wave>>2 (64 cols).
// Each wave: 2 A-frags x 4 B-frags = 8 WMMAs per K-step.
// Double-buffered LDS (one barrier per K-step); next K-tile prefetched into
// regs during compute; bf16 staging packed as ds_store_b64.
// Weights use non-temporal loads (streamed once; keep L2 for activations).
// If roi != nullptr: A element (m,k) = A[(m*14 + roi[k>>10])*1024 + (k&1023)]
// ---------------------------------------------------------------------------
__global__ __launch_bounds__(256)
void gemm_fast(const float* __restrict__ A, int lda,
               const float* __restrict__ W, int ldw,
               const float* __restrict__ bias,
               float* __restrict__ C, int ldc,
               int M, int N, int K, int act,
               const int* __restrict__ roi) {
  __shared__ unsigned short sA[2][128][40];
  __shared__ unsigned short sB[2][128][40];
  const int m_base = blockIdx.y * 128;
  const int n_base = blockIdx.x * 128;
  const int t    = threadIdx.x;
  const int lane = t & 31;
  const int wave = t >> 5;
  const int wm   = wave & 3;        // 32-row slice of M
  const int wn   = wave >> 2;       // 64-col slice of N
  const int half = lane >> 4;
  const int l16  = lane & 15;

  f32x8 acc[2][4];
  #pragma unroll
  for (int i = 0; i < 2; ++i)
    #pragma unroll
    for (int j = 0; j < 4; ++j)
      acc[i][j] = (f32x8){0.f,0.f,0.f,0.f,0.f,0.f,0.f,0.f};

  f32x4 ra[4], rb[4];

  auto load_tile = [&](int kk) {
    #pragma unroll
    for (int j = 0; j < 4; ++j) {
      int idx = t + 256 * j;
      int row = idx >> 3;          // 8 float4 per 32-wide row
      int c4  = (idx & 7) << 2;
      int gk  = kk + c4;
      if (roi) {
        const float* ap = &A[((size_t)(m_base + row) * 14 + roi[gk >> 10]) * 1024 + (gk & 1023)];
        ra[j] = *(const f32x4*)ap;
      } else {
        ra[j] = *(const f32x4*)&A[(size_t)(m_base + row) * lda + gk];
      }
      rb[j] = __builtin_nontemporal_load((const f32x4*)&W[(size_t)(n_base + row) * ldw + gk]);
    }
  };
  auto stage = [&](int bsel) {
    #pragma unroll
    for (int j = 0; j < 4; ++j) {
      int idx = t + 256 * j;
      int row = idx >> 3;
      int c4  = (idx & 7) << 2;
      u32x2 pa = { pack2bf(ra[j][0], ra[j][1]), pack2bf(ra[j][2], ra[j][3]) };
      u32x2 pb = { pack2bf(rb[j][0], rb[j][1]), pack2bf(rb[j][2], rb[j][3]) };
      *(u32x2*)&sA[bsel][row][c4] = pa;     // ds_store_b64 (8B aligned: pitch 80B, c4*2 in {0,8,..})
      *(u32x2*)&sB[bsel][row][c4] = pb;
    }
  };
  auto compute = [&](int bsel) {
    BFrag af[2];
    #pragma unroll
    for (int i = 0; i < 2; ++i) {
      int r = wm * 32 + i * 16 + l16;
      #pragma unroll
      for (int e = 0; e < 8; ++e) {
        af[i].u[e]     = sA[bsel][r][half * 8 + e];
        af[i].u[e + 8] = sA[bsel][r][half * 8 + 16 + e];
      }
    }
    #pragma unroll
    for (int j = 0; j < 4; ++j) {
      BFrag bf;
      int r = wn * 64 + j * 16 + l16;
      #pragma unroll
      for (int e = 0; e < 16; ++e) bf.u[e] = sB[bsel][r][half * 8 + (e & 7) + ((e >> 3) << 4)];
      acc[0][j] = __builtin_amdgcn_wmma_f32_16x16x32_bf16(false, af[0].v, false, bf.v,
                                                          (short)0, acc[0][j], false, false);
      acc[1][j] = __builtin_amdgcn_wmma_f32_16x16x32_bf16(false, af[1].v, false, bf.v,
                                                          (short)0, acc[1][j], false, false);
    }
  };

  load_tile(0);
  stage(0);
  __syncthreads();
  int buf = 0, kk = 0;
  for (;;) {
    int nxt = kk + 32;
    bool more = nxt < K;
    if (more) load_tile(nxt);     // prefetch next K-tile into regs during compute
    compute(buf);
    if (!more) break;
    stage(buf ^ 1);               // fill the other buffer (safe: consumed before last barrier)
    __syncthreads();              // single barrier per K-step
    buf ^= 1;
    kk = nxt;
  }

  #pragma unroll
  for (int i = 0; i < 2; ++i) {
    int mrow = m_base + wm * 32 + i * 16;
    #pragma unroll
    for (int j = 0; j < 4; ++j) {
      int n = n_base + wn * 64 + j * 16 + l16;
      float bv = bias[n];
      #pragma unroll
      for (int rr = 0; rr < 8; ++rr) {
        int m = mrow + rr + 8 * half;   // C/D layout: VGPR rr -> M=rr (+8 lanes 16-31)
        C[(size_t)m * ldc + n] = act_apply(acc[i][j][rr] + bv, act);
      }
    }
  }
}

// ---------------------------------------------------------------------------
// Edge GEMM (fully guarded, 64x64 tile) for N not multiple of 128 (N=12, N=80).
// ---------------------------------------------------------------------------
__global__ __launch_bounds__(256)
void gemm_edge(const float* __restrict__ A, int lda,
               const float* __restrict__ W, int ldw,
               const float* __restrict__ bias,
               float* __restrict__ C, int ldc,
               int M, int N, int K, int act) {
  __shared__ unsigned short sA[64][40];
  __shared__ unsigned short sB[64][40];
  const int m_base = blockIdx.y * 64;
  const int n_base = blockIdx.x * 64;
  const int t    = threadIdx.x;
  const int lane = t & 31;
  const int wave = t >> 5;
  const int wm   = wave & 3;
  const int wn   = wave >> 2;
  const int half = lane >> 4;
  const int l16  = lane & 15;

  f32x8 acc0 = {0.f,0.f,0.f,0.f,0.f,0.f,0.f,0.f};
  f32x8 acc1 = acc0;

  for (int kk = 0; kk < K; kk += 32) {
    #pragma unroll
    for (int j = 0; j < 8; ++j) {
      int i   = t + 256 * j;
      int row = i >> 5, col = i & 31;
      int gk  = kk + col;
      int gm  = m_base + row;
      float av = (gm < M && gk < K) ? A[(size_t)gm * lda + gk] : 0.f;
      sA[row][col] = f2bf(av);
      int gn = n_base + row;
      float wv = (gn < N && gk < K) ? W[(size_t)gn * ldw + gk] : 0.f;
      sB[row][col] = f2bf(wv);
    }
    __syncthreads();
    BFrag af, b0, b1;
    #pragma unroll
    for (int e = 0; e < 8; ++e) {
      af.u[e]     = sA[wm * 16 + l16][half * 8 + e];
      af.u[e + 8] = sA[wm * 16 + l16][half * 8 + 16 + e];
    }
    #pragma unroll
    for (int e = 0; e < 16; ++e) {
      b0.u[e] = sB[wn * 32 + l16][half * 8 + (e & 7) + ((e >> 3) << 4)];
      b1.u[e] = sB[wn * 32 + 16 + l16][half * 8 + (e & 7) + ((e >> 3) << 4)];
    }
    acc0 = __builtin_amdgcn_wmma_f32_16x16x32_bf16(false, af.v, false, b0.v,
                                                   (short)0, acc0, false, false);
    acc1 = __builtin_amdgcn_wmma_f32_16x16x32_bf16(false, af.v, false, b1.v,
                                                   (short)0, acc1, false, false);
    __syncthreads();
  }

  const int m0 = m_base + wm * 16;
  const int n0 = n_base + wn * 32;
  #pragma unroll
  for (int rr = 0; rr < 8; ++rr) {
    int m = m0 + rr + 8 * half;
    if (m >= M) continue;
    int n = n0 + l16;
    if (n < N) C[(size_t)m * ldc + n] = act_apply(acc0[rr] + bias[n], act);
    n = n0 + 16 + l16;
    if (n < N) C[(size_t)m * ldc + n] = act_apply(acc1[rr] + bias[n], act);
  }
}

// out[row] = LN(x[row] + r[row]) * g + b        (rows of 1024; r may be null)
__global__ __launch_bounds__(256)
void ln_k(const float* __restrict__ x, const float* __restrict__ r,
          const float* __restrict__ g, const float* __restrict__ be,
          float* __restrict__ out) {
  __shared__ float red[256];
  size_t base = (size_t)blockIdx.x * 1024;
  float v[4];
  float s = 0.f;
  #pragma unroll
  for (int j = 0; j < 4; ++j) {
    int c = threadIdx.x + 256 * j;
    float tv = x[base + c];
    if (r) tv += r[base + c];
    v[j] = tv; s += tv;
  }
  red[threadIdx.x] = s; __syncthreads();
  for (int st = 128; st > 0; st >>= 1) {
    if (threadIdx.x < st) red[threadIdx.x] += red[threadIdx.x + st];
    __syncthreads();
  }
  float mean = red[0] * (1.f / 1024.f);
  __syncthreads();
  float s2 = 0.f;
  #pragma unroll
  for (int j = 0; j < 4; ++j) { float d = v[j] - mean; s2 += d * d; }
  red[threadIdx.x] = s2; __syncthreads();
  for (int st = 128; st > 0; st >>= 1) {
    if (threadIdx.x < st) red[threadIdx.x] += red[threadIdx.x + st];
    __syncthreads();
  }
  float inv = rsqrtf(red[0] * (1.f / 1024.f) + 1e-5f);
  #pragma unroll
  for (int j = 0; j < 4; ++j) {
    int c = threadIdx.x + 256 * j;
    out[base + c] = (v[j] - mean) * inv * g[c] + be[c];
  }
}

// One thread per (q-token, batch, head): H=16, dh=64, B=256.
template <int NKV>
__global__ __launch_bounds__(256)
void attn_k(const float* __restrict__ qbuf, int sq, int qoff,
            const float* __restrict__ kvbuf, int skv, int koff, int voff,
            float* __restrict__ obuf, int so, int nq) {
  int tid = blockIdx.x * 256 + threadIdx.x;
  if (tid >= nq * 256 * 16) return;
  int h  = tid & 15;
  int b  = (tid >> 4) & 255;
  int tq = tid >> 12;
  const float* q = qbuf + ((size_t)(tq * 256 + b)) * sq + qoff + h * 64;
  float qr[64];
  #pragma unroll
  for (int i = 0; i < 64; ++i) qr[i] = q[i];
  float sc[NKV];
  float mx = -1e30f;
  #pragma unroll
  for (int s = 0; s < NKV; ++s) {
    const float* kp = kvbuf + ((size_t)(s * 256 + b)) * skv + koff + h * 64;
    float d = 0.f;
    #pragma unroll
    for (int i = 0; i < 64; ++i) d += qr[i] * kp[i];
    d *= 0.125f;                    // 1/sqrt(64)
    sc[s] = d; mx = fmaxf(mx, d);
  }
  float sum = 0.f;
  #pragma unroll
  for (int s = 0; s < NKV; ++s) { sc[s] = __expf(sc[s] - mx); sum += sc[s]; }
  float inv = 1.f / sum;
  float acc[64];
  #pragma unroll
  for (int i = 0; i < 64; ++i) acc[i] = 0.f;
  #pragma unroll
  for (int s = 0; s < NKV; ++s) {
    float w = sc[s] * inv;
    const float* vp = kvbuf + ((size_t)(s * 256 + b)) * skv + voff + h * 64;
    #pragma unroll
    for (int i = 0; i < 64; ++i) acc[i] += w * vp[i];
  }
  float* o = obuf + ((size_t)(tq * 256 + b)) * so + h * 64;
  #pragma unroll
  for (int i = 0; i < 64; ++i) o[i] = acc[i];
}

__global__ void gather_k(const float* __restrict__ x, float* __restrict__ out,
                         const int* __restrict__ idx, int ntok) {
  size_t i = (size_t)blockIdx.x * 256 + threadIdx.x;
  if (i >= (size_t)ntok * 256 * 1024) return;
  int d = i & 1023; int b = (i >> 10) & 255; int t = (int)(i >> 18);
  out[i] = x[(((size_t)idx[t] * 256) + b) * 1024 + d];
}

__global__ void scatter_k(const float* __restrict__ in, float* __restrict__ x,
                          const int* __restrict__ idx, int ntok) {
  size_t i = (size_t)blockIdx.x * 256 + threadIdx.x;
  if (i >= (size_t)ntok * 256 * 1024) return;
  int d = i & 1023; int b = (i >> 10) & 255; int t = (int)(i >> 18);
  x[(((size_t)idx[t] * 256) + b) * 1024 + d] = in[i];
}

__global__ void build_x_k(const float* __restrict__ emb, const float* __restrict__ ss,
                          const float* __restrict__ ms, const float* __restrict__ ls,
                          const float* __restrict__ a_ptr, float* __restrict__ x0) {
  size_t i = (size_t)blockIdx.x * 256 + threadIdx.x;
  if (i >= 24ull * 256 * 1024) return;
  int d = i & 1023; int b = (i >> 10) & 255; int s = (int)(i >> 18);
  float base;
  if      (s == 0) base = ss[d];
  else if (s == 1) base = ms[d];
  else if (s == 2) base = ls[d];
  else base = emb[((size_t)b * 21 + (s - 3)) * 1024 + d];
  int i2 = d & ~1;
  float dv  = __expf((float)i2 * (-9.210340371976184f / 1024.f)); // -ln(1e4)/D
  float ang = (float)s * dv;
  float pe  = (d & 1) ? __cosf(ang) : __sinf(ang);
  x0[i] = base + a_ptr[0] * pe;
}

__global__ void feat_init_k(const float* __restrict__ hi, const float* __restrict__ lo,
                            float* __restrict__ feat) {
  size_t i = (size_t)blockIdx.x * 256 + threadIdx.x;
  if (i >= 2ull * 256 * 1024) return;
  int d = i & 1023; int t = (int)(i >> 18);
  feat[i] = (t == 0) ? hi[d] : lo[d];
}

// out = a + g*o   (262144 elements)
__global__ void gate_combine_k(const float* __restrict__ a, const float* __restrict__ g,
                               const float* __restrict__ o, float* __restrict__ out) {
  size_t i = (size_t)blockIdx.x * 256 + threadIdx.x;
  if (i >= 262144ull) return;
  out[i] = a[i] + g[i] * o[i];
}

__global__ void concat_k(const float* __restrict__ zh, const float* __restrict__ zl,
                         float* __restrict__ cf) {
  size_t i = (size_t)blockIdx.x * 256 + threadIdx.x;
  if (i >= 524288ull) return;
  int d = (int)(i & 2047); int b = (int)(i >> 11);
  cf[i] = (d < 1024) ? zh[(size_t)b * 1024 + d] : zl[(size_t)b * 1024 + d - 1024];
}

__global__ void softmax12_k(const float* __restrict__ in, float* __restrict__ out) {
  int b = threadIdx.x;
  if (b >= 256) return;
  float v[12]; float mx = -1e30f;
  for (int j = 0; j < 12; ++j) { v[j] = in[b * 12 + j]; mx = fmaxf(mx, v[j]); }
  float s = 0.f;
  for (int j = 0; j < 12; ++j) { v[j] = __expf(v[j] - mx); s += v[j]; }
  float inv = 1.f / s;
  for (int j = 0; j < 12; ++j) out[b * 12 + j] = v[j] * inv;
}

__global__ void init_tables_k(int* __restrict__ w) {
  if (threadIdx.x != 0 || blockIdx.x != 0) return;
  w[0]=0; w[1]=1; w[2]=2; w[3]=3; w[4]=4;                 // dorsal @0
  for (int i = 0; i < 9; ++i) w[8+i] = 5 + i;             // ventral @8
  const int vis[7] = {5,0,6,1,7,2,8};
  for (int i = 0; i < 7; ++i) w[20+i] = vis[i];           // visual @20
  w[28]=9;  w[29]=10;                                     // faces @28
  w[30]=11; w[31]=12;                                     // words @30
  w[32]=3;                                                // places @32
  w[34]=4;  w[35]=13;                                     // bodies @34
  w[40]=0; w[41]=3; w[42]=4;                              // scale0 tokens @40
  w[44]=1; w[45]=5; w[46]=6; w[47]=7; w[48]=8; w[49]=9;   // scale1 @44
  w[52]=2; for (int i = 0; i < 14; ++i) w[53+i] = 10 + i; // scale2 @52
  w[68]=0; w[69]=1; w[70]=2;                              // inter @68
}

extern "C" void kernel_launch(void* const* d_in, const int* in_sizes, int n_in,
                              void* d_out, int out_size, void* d_ws, size_t ws_size,
                              hipStream_t stream) {
  (void)in_sizes; (void)n_in; (void)out_size; (void)ws_size;
  auto P = [&](int i) { return (const float*)d_in[i]; };

  int*   WSI = (int*)d_ws;
  float* F   = (float*)((char*)d_ws + 4096);
  const size_t SZ_X  = 24ull * 256 * 1024;   // 6,291,456
  const size_t SZ_T1 = 6144ull * 4096;       // 25,165,824
  float* X0 = F;
  float* Y  = X0 + SZ_X;
  float* T1 = Y + SZ_X;
  float* T2 = T1 + SZ_T1;
  float* QS = T1;                            // alias: qkv for token-subset MHAs
  float* AO = T1 + 12582912;                 // alias: attention output (subset MHAs)
  float* SX = T2;                            // alias: gathered tokens / proj output
  float* FEAT = T2 + SZ_X;
  float* FQ = FEAT + 524288;
  float* FO = FQ + 524288;
  float* FT = FO + 524288;
  float* FH = FT + 524288;                   // 512x4096
  float* G1 = FH + 2097152;
  float* GH = G1 + 262144;
  float* GL = GH + 262144;
  float* ZH = GL + 262144;
  float* ZL = ZH + 262144;
  float* CF = ZL + 262144;                   // 256x2048
  float* CH = CF + 524288;                   // 256x1024
  float* out = (float*)d_out;

  init_tables_k<<<1, 64, 0, stream>>>(WSI);

  auto gemm = [&](const float* A, int lda, const float* W, int ldw, const float* bias,
                  float* C, int ldc, int M, int N, int K, int act, const int* roi) {
    if ((M & 127) == 0 && (N & 127) == 0 && (K & 31) == 0) {
      dim3 g(N / 128, M / 128);
      gemm_fast<<<g, 256, 0, stream>>>(A, lda, W, ldw, bias, C, ldc, M, N, K, act, roi);
    } else {
      dim3 g((N + 63) / 64, (M + 63) / 64);
      gemm_edge<<<g, 256, 0, stream>>>(A, lda, W, ldw, bias, C, ldc, M, N, K, act);
    }
  };
  auto attn = [&](const float* qb, int sq, int qoff, const float* kvb, int skv,
                  int koff, int voff, float* ob, int so, int nq, int nkv) {
    dim3 g((nq * 256 * 16 + 255) / 256);
    switch (nkv) {
      case 3:  attn_k<3><<<g,256,0,stream>>>(qb,sq,qoff,kvb,skv,koff,voff,ob,so,nq); break;
      case 6:  attn_k<6><<<g,256,0,stream>>>(qb,sq,qoff,kvb,skv,koff,voff,ob,so,nq); break;
      case 15: attn_k<15><<<g,256,0,stream>>>(qb,sq,qoff,kvb,skv,koff,voff,ob,so,nq); break;
      default: attn_k<24><<<g,256,0,stream>>>(qb,sq,qoff,kvb,skv,koff,voff,ob,so,nq); break;
    }
  };

  // ---- Embedding front-end: raw emb (B,21,D) in T2, ldc = 21*1024 ----
  const float* src = P(0);
  gemm(src, 0, P(7),  5120, P(8),  T2 + 0*1024, 21504, 256, 1024, 5120, 0, WSI + 0);
  gemm(src, 0, P(9),  9216, P(10), T2 + 1*1024, 21504, 256, 1024, 9216, 0, WSI + 8);
  gemm(src, 0, P(11), 7168, P(12), T2 + 2*1024, 21504, 256, 1024, 7168, 0, WSI + 20);
  gemm(src, 0, P(13), 2048, P(14), T2 + 3*1024, 21504, 256, 1024, 2048, 0, WSI + 28);
  gemm(src, 0, P(15), 2048, P(16), T2 + 4*1024, 21504, 256, 1024, 2048, 0, WSI + 30);
  gemm(src, 0, P(17), 1024, P(18), T2 + 5*1024, 21504, 256, 1024, 1024, 0, WSI + 32);
  gemm(src, 0, P(19), 2048, P(20), T2 + 6*1024, 21504, 256, 1024, 2048, 0, WSI + 34);
  for (int r = 0; r < 14; ++r)   // per-ROI small linears
    gemm(src + r*1024, 14*1024, P(21) + (size_t)r*1024*1024, 1024, P(22) + r*1024,
         T2 + (size_t)(7 + r)*1024, 21504, 256, 1024, 1024, 0, nullptr);
  ln_k<<<256*21, 256, 0, stream>>>(T2, (const float*)nullptr, P(23), P(24), Y);
  build_x_k<<<(24*256*1024)/256, 256, 0, stream>>>(Y, P(1), P(2), P(3), P(6), X0);
  feat_init_k<<<(2*256*1024)/256, 256, 0, stream>>>(P(4), P(5), FEAT);

  auto msa_sub = [&](int ntok, const int* tokidx, int pb) {
    int Mr = ntok * 256;
    unsigned blocks = (unsigned)(((size_t)Mr * 1024 + 255) / 256);
    gather_k<<<blocks, 256, 0, stream>>>(Y, SX, tokidx, ntok);
    gemm(SX, 1024, P(pb), 1024, P(pb+1), QS, 3072, Mr, 3072, 1024, 0, nullptr);
    attn(QS, 3072, 0, QS, 3072, 1024, 2048, AO, 1024, ntok, ntok);
    gemm(AO, 1024, P(pb+2), 1024, P(pb+3), SX, 1024, Mr, 1024, 1024, 0, nullptr);
    scatter_k<<<blocks, 256, 0, stream>>>(SX, Y, tokidx, ntok);
  };

  // ---- Transformer layers ----
  for (int l = 0; l < 4; ++l) {
    int b0 = 25 + 40 * l;
    hipMemcpyAsync(Y, X0, SZ_X * sizeof(float), hipMemcpyDeviceToDevice, stream);
    msa_sub(3,  WSI + 40, b0 + 0);    // scale 0: [0,3,4]
    msa_sub(6,  WSI + 44, b0 + 4);    // scale 1
    msa_sub(15, WSI + 52, b0 + 8);    // scale 2
    msa_sub(3,  WSI + 68, b0 + 12);   // inter [0,1,2]
    // final full MHA over 24 tokens -> residual
    gemm(Y, 1024, P(b0+16), 1024, P(b0+17), T1, 3072, 6144, 3072, 1024, 0, nullptr);
    attn(T1, 3072, 0, T1, 3072, 1024, 2048, T2, 1024, 24, 24);
    gemm(T2, 1024, P(b0+18), 1024, P(b0+19), Y, 1024, 6144, 1024, 1024, 0, nullptr);
    ln_k<<<6144, 256, 0, stream>>>(X0, Y, P(b0+20), P(b0+21), X0);
    // FFN1 + LN
    gemm(X0, 1024, P(b0+22), 1024, P(b0+23), T1, 4096, 6144, 4096, 1024, 1, nullptr);
    gemm(T1, 4096, P(b0+24), 4096, P(b0+25), T2, 1024, 6144, 1024, 4096, 0, nullptr);
    ln_k<<<6144, 256, 0, stream>>>(X0, T2, P(b0+26), P(b0+27), X0);
    // Cross-attention: K,V from x (w_in rows D..3D contiguous), Q from feat
    gemm(X0, 1024, P(b0+28) + 1024*1024, 1024, P(b0+29) + 1024, T1, 2048,
         6144, 2048, 1024, 0, nullptr);
    gemm(FEAT, 1024, P(b0+28), 1024, P(b0+29), FQ, 1024, 512, 1024, 1024, 0, nullptr);
    attn(FQ, 1024, 0, T1, 2048, 0, 1024, FO, 1024, 2, 24);
    gemm(FO, 1024, P(b0+30), 1024, P(b0+31), FT, 1024, 512, 1024, 1024, 0, nullptr);
    ln_k<<<512, 256, 0, stream>>>(FEAT, FT, P(b0+32), P(b0+33), FEAT);
    // FFN2 + LN
    gemm(FEAT, 1024, P(b0+34), 1024, P(b0+35), FH, 4096, 512, 4096, 1024, 1, nullptr);
    gemm(FH, 4096, P(b0+36), 4096, P(b0+37), FT, 1024, 512, 1024, 4096, 0, nullptr);
    ln_k<<<512, 256, 0, stream>>>(FEAT, FT, P(b0+38), P(b0+39), FEAT);
  }

  // ---- Heads ----
  const float* zh = FEAT;
  const float* zl = FEAT + 256 * 1024;
  gemm(zh, 1024, P(185), 1024, P(186), G1, 1024, 256, 1024, 1024, 1, nullptr);
  gemm(G1, 1024, P(187), 1024, P(188), GH, 1024, 256, 1024, 1024, 2, nullptr);
  gemm(zl, 1024, P(185), 1024, P(186), G1, 1024, 256, 1024, 1024, 1, nullptr);
  gemm(G1, 1024, P(187), 1024, P(188), GL, 1024, 256, 1024, 1024, 2, nullptr);
  gate_combine_k<<<1024, 256, 0, stream>>>(zh, GH, zl, ZH);
  gate_combine_k<<<1024, 256, 0, stream>>>(zl, GL, zh, ZL);
  gemm(ZH, 1024, P(189), 1024, P(190), out,          768, 256, 768, 1024, 0, nullptr);
  gemm(ZL, 1024, P(191), 1024, P(192), out + 196608, 768, 256, 768, 1024, 0, nullptr);
  concat_k<<<2048, 256, 0, stream>>>(ZH, ZL, CF);
  gemm(CF, 2048, P(193), 2048, P(194), CH, 1024, 256, 1024, 2048, 1, nullptr);
  gemm(CH, 1024, P(195), 1024, P(196), out + 393216, 12, 256, 12, 1024, 0, nullptr);
  softmax12_k<<<1, 256, 0, stream>>>(out + 393216, out + 396288);
  gemm(CF, 2048, P(197), 2048, P(198), CH, 1024, 256, 1024, 2048, 1, nullptr);
  gemm(CH, 1024, P(199), 1024, P(200), out + 399360, 80, 256, 80, 1024, 2, nullptr);
}
